// SelfAttention_54906861912354
// MI455X (gfx1250) — compile-verified, matrix operand order checked by reading, not run on hardware
//
#include <hip/hip_runtime.h>

#define DEVI __device__ __forceinline__

typedef __attribute__((ext_vector_type(16))) _Float16 v16h;
typedef __attribute__((ext_vector_type(8)))  _Float16 v8h;
typedef __attribute__((ext_vector_type(8)))  float    v8f;
typedef __attribute__((ext_vector_type(4)))  float    v4f;
typedef __attribute__((ext_vector_type(4)))  unsigned int v4u;
typedef __attribute__((ext_vector_type(8)))  int      v8i;
typedef __attribute__((ext_vector_type(4)))  int      v4i;

constexpr int BB = 4;
constexpr int S  = 4096;
constexpr int H  = 1024;
constexpr float SCALE = 0.03125f;   // 1/sqrt(1024)

// Staged-chunk geometry: 256 f16 columns per row, 16-byte pad per row so the
// LDS row stride (528 B = 132 dwords == 4 banks mod 64) spreads the 16-lane
// b128 fragment reads across all 64 banks (conflict-free).
constexpr int CHW  = 256;           // chunk width in f16 elements (512 B row)
constexpr int ROWF = CHW + 8;       // f16 per LDS row incl. 16-B pad

#if defined(__has_builtin)
#  if __has_builtin(__builtin_amdgcn_tensor_load_to_lds)
#    define TDM_OK 1
#  endif
#endif
#ifndef TDM_OK
#  define TDM_OK 0
#endif
#if __has_include(<hip/amd_detail/amd_gfx1250_TDM.h>)
#  define TDM_6ARG 1
#else
#  define TDM_6ARG 0
#endif

DEVI v8f vzero8() { v8f z = {0.f,0.f,0.f,0.f,0.f,0.f,0.f,0.f}; return z; }

DEVI v8f wmma16(v16h a, v16h b, v8f c) {
  return __builtin_amdgcn_wmma_f32_16x16x32_f16(
      /*neg_a=*/false, a, /*neg_b=*/false, b,
      /*c_mod=*/(short)0, c, /*reuse_a=*/false, /*reuse_b=*/false);
}

// 16x32 f16 A/B fragment from row-major f16 global memory.
DEVI v16h load_frag_f16(const _Float16* __restrict__ p, int ld, long row, int k0, int lane) {
  int r = lane & 15, hi = lane >> 4;
  const _Float16* base = p + (size_t)(row + r) * ld + k0 + hi * 8;
  v8h lo  = *(const v8h*)(base);
  v8h hi8 = *(const v8h*)(base + 16);
  v16h out;
#pragma unroll
  for (int i = 0; i < 8; ++i) { out[i] = lo[i]; out[i + 8] = hi8[i]; }
  return out;
}

// Same fragment sourced from fp32 memory with inline convert.
DEVI v16h load_frag_f32(const float* __restrict__ p, int ld, long row, int k0, int lane) {
  int r = lane & 15, hi = lane >> 4;
  const float* base = p + (size_t)(row + r) * ld + k0 + hi * 8;
  v4f a0 = *(const v4f*)(base);
  v4f a1 = *(const v4f*)(base + 4);
  v4f b0 = *(const v4f*)(base + 16);
  v4f b1 = *(const v4f*)(base + 20);
  v16h out;
#pragma unroll
  for (int i = 0; i < 4; ++i) {
    out[i]      = (_Float16)a0[i];
    out[4 + i]  = (_Float16)a1[i];
    out[8 + i]  = (_Float16)b0[i];
    out[12 + i] = (_Float16)b1[i];
  }
  return out;
}

// 16x32 fragment from a staged LDS tile (row stride ROWF f16, rows padded).
DEVI v16h load_frag_lds(const _Float16* buf, int row0, int k0, int lane) {
  int r = lane & 15, hi = lane >> 4;
  const _Float16* base = buf + (row0 + r) * ROWF + k0 + hi * 8;
  v8h lo  = *(const v8h*)(base);
  v8h hi8 = *(const v8h*)(base + 16);
  v16h out;
#pragma unroll
  for (int i = 0; i < 8; ++i) { out[i] = lo[i]; out[i + 8] = hi8[i]; }
  return out;
}

// ---------------------------------------------------------------------------
// TDM staging: async DMA of a [tileRows x CHW] f16 tile (global row stride
// strideElems) into LDS with 16-B row padding. Issued by wave 0 only;
// completion tracked with TENSORcnt. Fallback: cooperative vector copy.
// ---------------------------------------------------------------------------
DEVI void stage_issue(_Float16* ldsDst, const _Float16* gsrc,
                      int tileRows, int strideElems, int tid) {
#if TDM_OK
  if (tid < 32) {
    unsigned long long ga = (unsigned long long)(size_t)gsrc;
    unsigned ldsOff = (unsigned)(size_t)(void*)ldsDst;
    v4u g0;
    g0[0] = 1u;                                   // count=1, user descriptor
    g0[1] = ldsOff;                               // lds_addr
    g0[2] = (unsigned)ga;                         // global_addr[31:0]
    g0[3] = (unsigned)((ga >> 32) & 0x01FFFFFFu)  // global_addr[56:32]
            | 0x80000000u;                        // type=2 ("image")
    v8i g1;
    g1[0] = (int)((1u << 16)      // data_size = 2 bytes
                | (1u << 20)      // pad_enable
                | (6u << 22)      // pad_interval = 128 dwords (512 B = 1 row)
                | (3u << 25));    // pad_amount  = 4 dwords (16 B)
    unsigned tenW = (unsigned)strideElems;        // tensor_dim0 (>= tile width)
    unsigned tenH = 1u << 20;                     // tensor_dim1 (big; tile in range)
    g1[1] = (int)((tenW & 0xFFFFu) << 16);                        // dim0 lo16
    g1[2] = (int)((tenW >> 16) | ((tenH & 0xFFFFu) << 16));       // dim0 hi / dim1 lo
    g1[3] = (int)(((tenH >> 16) & 0xFFFFu) | ((unsigned)CHW << 16)); // dim1 hi / tile_dim0
    g1[4] = (int)(tileRows & 0xFFFF);             // tile_dim1 (tile_dim2 = 0)
    g1[5] = (int)strideElems;                     // tensor_dim0_stride lo32
    g1[6] = 0;
    g1[7] = 0;
    v4i g2 = {0, 0, 0, 0};
    v4i g3 = {0, 0, 0, 0};
#if TDM_6ARG
    v8i g4 = {0, 0, 0, 0, 0, 0, 0, 0};
    __builtin_amdgcn_tensor_load_to_lds(g0, g1, g2, g3, g4, 0);
#else
    __builtin_amdgcn_tensor_load_to_lds(g0, g1, g2, g3, 0);
#endif
  }
  asm volatile("" ::: "memory");   // keep LDS reads ordered vs. the DMA
#else
  const int nch = tileRows * (CHW / 8);
  for (int c = tid; c < nch; c += 256) {
    int row = c / (CHW / 8);
    int col = (c % (CHW / 8)) * 8;
    *(v8h*)(ldsDst + row * ROWF + col) =
        *(const v8h*)(gsrc + (size_t)row * strideElems + col);
  }
#endif
}

DEVI void stage_wait(int allowOutstanding, int tid) {
#if TDM_OK
  if (tid < 32) {
#if defined(__has_builtin) && __has_builtin(__builtin_amdgcn_s_wait_tensorcnt)
    if (allowOutstanding) __builtin_amdgcn_s_wait_tensorcnt(1);
    else                  __builtin_amdgcn_s_wait_tensorcnt(0);
#else
    if (allowOutstanding) asm volatile("s_wait_tensorcnt 0x1" ::: "memory");
    else                  asm volatile("s_wait_tensorcnt 0x0" ::: "memory");
#endif
  }
  asm volatile("" ::: "memory");
#else
  (void)allowOutstanding; (void)tid;
#endif
}

// ---------------------------------------------------------------------------
// Stage 1: q = (x@Wq^T + bq)*scale (f16), k = x@Wk^T + bk (f16),
//          vT[b][h][s] = (x@Wv^T + bv)[b][s][h] (f16, transposed store).
// One wave per 16x64 output tile of one matrix.
// ---------------------------------------------------------------------------
__global__ __launch_bounds__(256) void sa_qkv_proj(
    const float* __restrict__ x,
    const float* __restrict__ Wq, const float* __restrict__ bq,
    const float* __restrict__ Wk, const float* __restrict__ bk,
    const float* __restrict__ Wv, const float* __restrict__ bv,
    _Float16* __restrict__ qh, _Float16* __restrict__ kh,
    _Float16* __restrict__ vT)
{
  const int lane = threadIdx.x & 31;
  const int wid  = blockIdx.x * (blockDim.x >> 5) + (threadIdx.x >> 5);
  const int MT = (BB * S) / 16;
  const int NC = H / 64;
  const int mat = wid / (MT * NC);
  const int rem = wid % (MT * NC);
  const long m0 = (long)(rem / NC) * 16;
  const int  n0 = (rem % NC) * 64;

  const float* W    = (mat == 0) ? Wq : (mat == 1) ? Wk : Wv;
  const float* bias = (mat == 0) ? bq : (mat == 1) ? bk : bv;

  v8f acc[4];
#pragma unroll
  for (int j = 0; j < 4; ++j) acc[j] = vzero8();

  for (int k0 = 0; k0 < H; k0 += 32) {
    v16h a = load_frag_f32(x, H, m0, k0, lane);
#pragma unroll
    for (int j = 0; j < 4; ++j) {
      v16h bf = load_frag_f32(W, H, n0 + j * 16, k0, lane);
      acc[j] = wmma16(a, bf, acc[j]);
    }
  }

  const int r16 = lane & 15, hi = lane >> 4;
  if (mat == 2) {
    const int b  = (int)(m0 / S);
    const int s0 = (int)(m0 - (long)b * S) + hi * 8;
#pragma unroll
    for (int j = 0; j < 4; ++j) {
      const int n = n0 + j * 16 + r16;
      const float bv_ = bias[n];
      v8h pack;
#pragma unroll
      for (int r = 0; r < 8; ++r) pack[r] = (_Float16)(acc[j][r] + bv_);
      *(v8h*)(vT + ((size_t)b * H + n) * S + s0) = pack;
    }
  } else {
    _Float16* dst = (mat == 0) ? qh : kh;
    const float sc = (mat == 0) ? SCALE : 1.0f;
#pragma unroll
    for (int j = 0; j < 4; ++j) {
      const int n = n0 + j * 16 + r16;
      const float bv_ = bias[n];
#pragma unroll
      for (int r = 0; r < 8; ++r) {
        const long m = m0 + r + hi * 8;
        dst[(size_t)m * H + n] = (_Float16)((acc[j][r] + bv_) * sc);
      }
    }
  }
}

// ---------------------------------------------------------------------------
// Stage 2: S[b,q,k] = qhat . k (f16 store) + per-row running max.
// Block: 8 waves x 16 q-rows. The 32-row k-tile (shared by all waves) is
// staged into LDS by the TDM in CHW-wide chunks, double buffered.
// ---------------------------------------------------------------------------
__global__ __launch_bounds__(256) void sa_scores(
    const _Float16* __restrict__ qh, const _Float16* __restrict__ kh,
    _Float16* __restrict__ sb, float* __restrict__ rowmax)
{
  __shared__ _Float16 kbuf[2][32 * ROWF];   // 2 x 16.5 KB

  const int tid  = threadIdx.x;
  const int lane = tid & 31;
  const int wave = tid >> 5;
  const int b    = blockIdx.y;
  const int m0l  = blockIdx.x * 128 + wave * 16;
  const long rowq = (long)b * S + m0l;
  const int r16 = lane & 15, hi = lane >> 4;

  float mx[8];
#pragma unroll
  for (int r = 0; r < 8; ++r) mx[r] = -3.0e38f;

  const int NH  = H / CHW;                  // chunks per k-tile (4)
  const int NIT = (S / 32) * NH;            // total staged chunks (512)

  // prologue: stage chunk 0 (kt=0, hc=0)
  stage_issue(kbuf[0], kh + (size_t)b * S * H, 32, H, tid);

  int it = 0;
  for (int kt = 0; kt < S; kt += 32) {
    v8f acc0 = vzero8(), acc1 = vzero8();
    for (int hcI = 0; hcI < NH; ++hcI, ++it) {
      const int nxt = it + 1;
      if (nxt < NIT) {
        const int nkt = (nxt / NH) * 32;
        const int nhc = (nxt % NH) * CHW;
        stage_issue(kbuf[nxt & 1],
                    kh + ((size_t)b * S + nkt) * H + nhc, 32, H, tid);
      }
      stage_wait(nxt < NIT, tid);
      __syncthreads();
      const _Float16* buf = kbuf[it & 1];
      const int hbase = hcI * CHW;
      for (int h0 = 0; h0 < CHW; h0 += 32) {
        v16h a  = load_frag_f16(qh, H, rowq, hbase + h0, lane);
        v16h b0 = load_frag_lds(buf, 0,  h0, lane);
        v16h b1 = load_frag_lds(buf, 16, h0, lane);
        acc0 = wmma16(a, b0, acc0);
        acc1 = wmma16(a, b1, acc1);
      }
      __syncthreads();
    }
#pragma unroll
    for (int r = 0; r < 8; ++r) {
      const float v0 = acc0[r], v1 = acc1[r];
      mx[r] = fmaxf(mx[r], fmaxf(v0, v1));
      const size_t rowoff = ((size_t)b * S + m0l + r + hi * 8) * (size_t)S;
      sb[rowoff + kt + r16]      = (_Float16)v0;
      sb[rowoff + kt + 16 + r16] = (_Float16)v1;
    }
  }

  // reduce max across the 16 lanes of each half-wave (C layout halves)
#pragma unroll
  for (int r = 0; r < 8; ++r) {
#pragma unroll
    for (int d = 1; d < 16; d <<= 1)
      mx[r] = fmaxf(mx[r], __shfl_xor(mx[r], d, 32));
  }
  if (r16 == 0) {
#pragma unroll
    for (int r = 0; r < 8; ++r)
      rowmax[(size_t)b * S + m0l + hi * 8 + r] = mx[r];
  }
}

// ---------------------------------------------------------------------------
// Stage 3: in-place E = exp(S - rowmax), rowsum = sum_k E. One wave per row.
// ---------------------------------------------------------------------------
__global__ __launch_bounds__(256) void sa_softexp(
    _Float16* __restrict__ sb, const float* __restrict__ rowmax,
    float* __restrict__ rowsum)
{
  const int lane = threadIdx.x & 31;
  const int wave = threadIdx.x >> 5;
  const long row = (long)blockIdx.x * 8 + wave;
  const float m = rowmax[row];
  _Float16* p = sb + (size_t)row * S;
  float sum = 0.f;
  for (int i = 0; i < S / 256; ++i) {
    v8h vv = *(const v8h*)(p + i * 256 + lane * 8);
    v8h ee;
#pragma unroll
    for (int j = 0; j < 8; ++j) {
      const float e = __expf((float)vv[j] - m);
      sum += e;
      ee[j] = (_Float16)e;
    }
    *(v8h*)(p + i * 256 + lane * 8) = ee;
  }
#pragma unroll
  for (int d = 1; d < 32; d <<= 1) sum += __shfl_xor(sum, d, 32);
  if (lane == 0) rowsum[row] = sum;
}

// ---------------------------------------------------------------------------
// Stage 4: out[b,q,h] = (E @ v)[q,h] / rowsum[q], fp32 output.
// Block: one 16-row q-tile; wave w covers h-chunk [w*128, w*128+128).
// The shared 16-row E tile (A operand) is TDM-staged into LDS, double buffered;
// per-wave vT fragments stream from global.
// ---------------------------------------------------------------------------
__global__ __launch_bounds__(256) void sa_pv(
    const _Float16* __restrict__ eb, const _Float16* __restrict__ vT,
    const float* __restrict__ rowsum, float* __restrict__ out)
{
  __shared__ _Float16 ebuf[2][16 * ROWF];   // 2 x 8.25 KB

  const int tid  = threadIdx.x;
  const int lane = tid & 31;
  const int wave = tid >> 5;
  const long m0  = (long)blockIdx.x * 16;   // global q row in [0, B*S)
  const int  h0  = wave * 128;
  const int  b   = (int)(m0 / S);

  v8f acc[8];
#pragma unroll
  for (int j = 0; j < 8; ++j) acc[j] = vzero8();

  const _Float16* vTb = vT + (size_t)b * H * S;
  const int NIT = S / CHW;                  // 16 staged chunks

  stage_issue(ebuf[0], eb + (size_t)m0 * S, 16, S, tid);

  for (int it = 0; it < NIT; ++it) {
    const int nxt = it + 1;
    if (nxt < NIT) {
      stage_issue(ebuf[nxt & 1], eb + (size_t)m0 * S + nxt * CHW, 16, S, tid);
    }
    stage_wait(nxt < NIT, tid);
    __syncthreads();
    const _Float16* buf = ebuf[it & 1];
    const int kbase = it * CHW;
    for (int k0 = 0; k0 < CHW; k0 += 32) {
      v16h a = load_frag_lds(buf, 0, k0, lane);
#pragma unroll
      for (int j = 0; j < 8; ++j) {
        v16h bf = load_frag_f16(vTb, S, h0 + j * 16, kbase + k0, lane);
        acc[j] = wmma16(a, bf, acc[j]);
      }
    }
    __syncthreads();
  }

  const int r16 = lane & 15, hi = lane >> 4;
  float inv[8];
#pragma unroll
  for (int r = 0; r < 8; ++r) inv[r] = 1.0f / rowsum[m0 + r + hi * 8];
#pragma unroll
  for (int j = 0; j < 8; ++j) {
#pragma unroll
    for (int r = 0; r < 8; ++r) {
      out[(size_t)(m0 + r + hi * 8) * H + h0 + j * 16 + r16] = acc[j][r] * inv[r];
    }
  }
}

// ---------------------------------------------------------------------------
extern "C" void kernel_launch(void* const* d_in, const int* in_sizes, int n_in,
                              void* d_out, int out_size, void* d_ws, size_t ws_size,
                              hipStream_t stream) {
  (void)in_sizes; (void)n_in; (void)out_size; (void)ws_size;
  const float* x  = (const float*)d_in[0];
  const float* Wq = (const float*)d_in[1];
  const float* bq = (const float*)d_in[2];
  const float* Wk = (const float*)d_in[3];
  const float* bk = (const float*)d_in[4];
  const float* Wv = (const float*)d_in[5];
  const float* bv = (const float*)d_in[6];
  float* out = (float*)d_out;

  char* ws = (char*)d_ws;
  const size_t qkvElems = (size_t)BB * S * H;
  _Float16* qh = (_Float16*)ws;  ws += qkvElems * sizeof(_Float16);
  _Float16* kh = (_Float16*)ws;  ws += qkvElems * sizeof(_Float16);
  _Float16* vT = (_Float16*)ws;  ws += qkvElems * sizeof(_Float16);
  _Float16* sb = (_Float16*)ws;  ws += (size_t)BB * S * S * sizeof(_Float16);
  float* rowmax = (float*)ws;    ws += (size_t)BB * S * sizeof(float);
  float* rowsum = (float*)ws;

  {
    const int waves = (BB * S / 16) * (H / 64) * 3;
    sa_qkv_proj<<<waves / 8, 256, 0, stream>>>(x, Wq, bq, Wk, bk, Wv, bv, qh, kh, vT);
  }
  {
    dim3 grid(S / 128, BB);
    sa_scores<<<grid, 256, 0, stream>>>(qh, kh, sb, rowmax);
  }
  {
    sa_softexp<<<(BB * S) / 8, 256, 0, stream>>>(sb, rowmax, rowsum);
  }
  {
    sa_pv<<<BB * S / 16, 256, 0, stream>>>(sb, vT, rowsum, out);
  }
}
